// WalkKANMulti_30073361007192
// MI455X (gfx1250) — matrix operand order repaired and checked by reading, not compiled, not run on hardware
//
#include <hip/hip_runtime.h>
#include <hip/hip_bf16.h>
#include <math.h>

// ---------------------------------------------------------------------------
// KAN pipeline: precomputed bf16 feature expansion + pure bf16 WMMA GEMMs
// (gfx1250 wave32, v_wmma_f32_16x16x32_bf16, async global->LDS staging)
// ---------------------------------------------------------------------------

typedef __attribute__((ext_vector_type(16))) __bf16        v16bf;
typedef __attribute__((ext_vector_type(8)))  float         v8f;
typedef __attribute__((ext_vector_type(8)))  unsigned int  v8u;
typedef __attribute__((ext_vector_type(4)))  int           v4i;

#define AS1 __attribute__((address_space(1)))
#define AS3 __attribute__((address_space(3)))

#define NCOEF 8     // B-spline basis count (grid=5, k=3)
#define FPAD  16    // padded feature slots per input (8 basis + silu + 7 zeros)
#define BSZ   4096
#define MT    128   // M tile (batch rows)
#define NT    128   // N tile (output cols)
#define KTT   64    // K tile staged in LDS (2 WMMA K-steps of 32)

#if __has_builtin(__builtin_amdgcn_global_load_async_to_lds_b128)
#define USE_ASYNC_LDS 1
#else
#define USE_ASYNC_LDS 0
#endif

__device__ __forceinline__ unsigned short f2bf(float f) {
  unsigned int u = __builtin_bit_cast(unsigned int, f);
  u += 0x7FFFu + ((u >> 16) & 1u);          // round-to-nearest-even
  return (unsigned short)(u >> 16);
}

// 16 features of one scalar x: [B0..B7, silu(x), 0 x7]
// Uniform extended grid: knots t_j = (j-3)*0.4 - 1, j = 0..11 (Cox-de Boor).
__device__ __forceinline__ void kan_feats(float x, float* f) {
  float Bv[11];
#pragma unroll
  for (int j = 0; j < 11; ++j) {
    float t0 = (float)(j - 3) * 0.4f - 1.0f;
    float t1 = (float)(j - 2) * 0.4f - 1.0f;
    Bv[j] = (x >= t0 && x < t1) ? 1.0f : 0.0f;
  }
#pragma unroll
  for (int p = 1; p <= 3; ++p) {
    float inv = 1.0f / (0.4f * (float)p);
#pragma unroll
    for (int j = 0; j < 11 - p; ++j) {
      float tj   = (float)(j - 3) * 0.4f - 1.0f;                      // t_j
      float tj1p = (float)(j - 2) * 0.4f - 1.0f + 0.4f * (float)p;    // t_{j+p+1}
      Bv[j] = (x - tj) * inv * Bv[j] + (tj1p - x) * inv * Bv[j + 1];
    }
  }
#pragma unroll
  for (int j = 0; j < NCOEF; ++j) f[j] = Bv[j];
  f[8] = x / (1.0f + __expf(-x));                                     // silu
#pragma unroll
  for (int j = NCOEF + 1; j < FPAD; ++j) f[j] = 0.0f;
}

// X(B,IN) fp32 -> F(B, IN*16) bf16 feature matrix (done ONCE per layer).
__global__ void expand_feats(const float* __restrict__ X, unsigned short* __restrict__ F,
                             long total /* B*IN */) {
  long idx = (long)blockIdx.x * blockDim.x + threadIdx.x;
  if (idx >= total) return;
  float f[FPAD];
  kan_feats(X[idx], f);
  unsigned int u[8];
#pragma unroll
  for (int j = 0; j < 8; ++j)
    u[j] = (unsigned int)f2bf(f[2 * j]) | ((unsigned int)f2bf(f[2 * j + 1]) << 16);
  uint4* dst = (uint4*)(F + idx * FPAD);
  dst[0] = make_uint4(u[0], u[1], u[2], u[3]);
  dst[1] = make_uint4(u[4], u[5], u[6], u[7]);
}

// Fuse coef*ss and sb into bf16 W(K16=IN*FPAD, OUT), row-major, zero pad rows.
__global__ void pack_w(const float* __restrict__ coef, const float* __restrict__ sb,
                       const float* __restrict__ ss, unsigned short* __restrict__ W,
                       int IN, int OUT) {
  long idx = (long)blockIdx.x * blockDim.x + threadIdx.x;
  if (idx >= (long)IN * OUT) return;
  int i = (int)(idx / OUT), o = (int)(idx % OUT);
  float s = ss[idx];
  const float* c = coef + idx * NCOEF;
  unsigned short* w = W + (long)i * FPAD * OUT + o;
#pragma unroll
  for (int g = 0; g < NCOEF; ++g) w[(long)g * OUT] = f2bf(c[g] * s);
  w[(long)NCOEF * OUT] = f2bf(sb[idx]);
#pragma unroll
  for (int g = NCOEF + 1; g < FPAD; ++g) w[(long)g * OUT] = 0;   // bf16 +0.0
}

// Y(BSZ,OUT) = F(BSZ,K16) @ W(K16,OUT)  [+ R]  — pure bf16 WMMA, f32 acc
__global__ __launch_bounds__(256)
void kan_gemm(const unsigned short* __restrict__ F, const unsigned short* __restrict__ W,
              float* __restrict__ Y, const float* __restrict__ R,
              int K16, int OUT) {
  __shared__ unsigned short ldsA[MT * KTT];   // [row][k]   features
  __shared__ unsigned short ldsB[NT * KTT];   // [col][k]   weights transposed

  const int tid  = threadIdx.x;
  const int lane = tid & 31;
  const int wave = tid >> 5;
  const int wm   = wave >> 2;                 // 0..1  -> 64 rows each
  const int wn   = wave & 3;                  // 0..3  -> 32 cols each
  const int m0   = blockIdx.x * MT;
  const int n0   = blockIdx.y * NT;
  const bool fullN = (n0 + NT <= OUT);

  v8f acc[4][2];
#pragma unroll
  for (int mi = 0; mi < 4; ++mi)
#pragma unroll
    for (int ni = 0; ni < 2; ++ni)
      acc[mi][ni] = (v8f){0.f, 0.f, 0.f, 0.f, 0.f, 0.f, 0.f, 0.f};

  const int half = lane >> 4;
  const int l16  = lane & 15;

  for (int kb = 0; kb < K16; kb += KTT) {
    __syncthreads();
    { // ---- A tile: 128 rows x 64 bf16 ----
      int row = tid >> 1, seg = tid & 1;      // 64B per thread
      const unsigned short* gp = F + (long)(m0 + row) * K16 + kb + seg * 32;
      unsigned short* lp = &ldsA[row * KTT + seg * 32];
#if USE_ASYNC_LDS
      // CDNA5 async DMA straight into LDS (ASYNCcnt), no VGPR staging
      AS1 v4i* g = (AS1 v4i*)gp;
      AS3 v4i* l = (AS3 v4i*)(unsigned int)(unsigned long long)lp;
      __builtin_amdgcn_global_load_async_to_lds_b128(g, l, 0, 0);
      __builtin_amdgcn_global_load_async_to_lds_b128(g, l, 16, 0);
      __builtin_amdgcn_global_load_async_to_lds_b128(g, l, 32, 0);
      __builtin_amdgcn_global_load_async_to_lds_b128(g, l, 48, 0);
#else
      const uint4* s = (const uint4*)gp;
      uint4 t0 = s[0], t1 = s[1], t2 = s[2], t3 = s[3];   // issue all loads first
      uint4* d = (uint4*)lp;
      d[0] = t0; d[1] = t1; d[2] = t2; d[3] = t3;
#endif
    }
    { // ---- B tile: 64 x 128 of W, stored transposed [col][k] ----
      // Each thread: 4 rows x 8 cols sub-block -> 4 coalesced uint4 loads,
      // then 8 packed b64 LDS stores (4 consecutive K per column).
      int rg = tid >> 4, cg = tid & 15;
      int r0 = rg * 4, c0 = cg * 8;
      const unsigned short* src = W + (long)(kb + r0) * OUT + n0 + c0;
      if (kb + KTT < K16)                     // global_prefetch next K slab
        __builtin_prefetch(W + (long)(kb + KTT + r0) * OUT + n0 + c0, 0, 0);
      if (fullN) {
        uint4 q0 = *(const uint4*)(src);
        uint4 q1 = *(const uint4*)(src + OUT);
        uint4 q2 = *(const uint4*)(src + 2 * OUT);
        uint4 q3 = *(const uint4*)(src + 3 * OUT);
        const unsigned short* p0 = (const unsigned short*)&q0;
        const unsigned short* p1 = (const unsigned short*)&q1;
        const unsigned short* p2 = (const unsigned short*)&q2;
        const unsigned short* p3 = (const unsigned short*)&q3;
#pragma unroll
        for (int c = 0; c < 8; ++c) {
          uint2 v;
          v.x = (unsigned int)p0[c] | ((unsigned int)p1[c] << 16);
          v.y = (unsigned int)p2[c] | ((unsigned int)p3[c] << 16);
          *(uint2*)&ldsB[(c0 + c) * KTT + r0] = v;
        }
      } else {                                // guarded path (OUT=64 layer only)
#pragma unroll
        for (int j = 0; j < 4; ++j)
#pragma unroll
          for (int c = 0; c < 8; ++c)
            ldsB[(c0 + c) * KTT + r0 + j] =
                (n0 + c0 + c < OUT) ? W[(long)(kb + r0 + j) * OUT + n0 + c0 + c]
                                    : (unsigned short)0;
      }
    }
#if USE_ASYNC_LDS
#if __has_builtin(__builtin_amdgcn_s_wait_asynccnt)
    __builtin_amdgcn_s_wait_asynccnt(0);
#else
    asm volatile("s_wait_asynccnt 0" ::: "memory");
#endif
#endif
    __syncthreads();

    const unsigned int* A32 = (const unsigned int*)ldsA;
    const unsigned int* B32 = (const unsigned int*)ldsB;

#pragma unroll
    for (int ks = 0; ks < 2; ++ks) {          // two WMMA K-steps per LDS tile
      // B fragments: lane l16 = N; VGPR0-3 K={0..7|8..15}, VGPR4-7 K={16..23|24..31}
      v16bf bf[2];
#pragma unroll
      for (int ni = 0; ni < 2; ++ni) {
        int col = wn * 32 + ni * 16 + l16;
        v8u t;
#pragma unroll
        for (int v = 0; v < 8; ++v)
          t[v] = B32[col * (KTT / 2) + ks * 16 + half * 4 + v + (v >> 2) * 4];
        bf[ni] = __builtin_bit_cast(v16bf, t);
      }
      // A fragments: lane l16 = M; VGPR0-3 K={0..7|8..15}, VGPR4-7 K={16..23|24..31}
#pragma unroll
      for (int mi = 0; mi < 4; ++mi) {
        int row = wm * 64 + mi * 16 + l16;
        v8u t;
#pragma unroll
        for (int v = 0; v < 8; ++v) {
          int K = (v < 4) ? (8 * half + 2 * v) : (16 + 8 * half + 2 * (v - 4));
          t[v] = A32[row * (KTT / 2) + ks * 16 + (K >> 1)];
        }
        v16bf af = __builtin_bit_cast(v16bf, t);
#pragma unroll
        for (int ni = 0; ni < 2; ++ni)
          acc[mi][ni] = __builtin_amdgcn_wmma_f32_16x16x32_bf16(
              false, af, false, bf[ni], (short)0, acc[mi][ni], false, false);
      }
    }
  }

  // ---- epilogue: C/D layout VGPR r -> M = r + 8*half, N = l16 ----
#pragma unroll
  for (int mi = 0; mi < 4; ++mi)
#pragma unroll
    for (int ni = 0; ni < 2; ++ni) {
      int col = n0 + wn * 32 + ni * 16 + l16;
      if (col < OUT) {
#pragma unroll
        for (int r = 0; r < 8; ++r) {
          int row = m0 + wm * 64 + mi * 16 + r + 8 * half;
          float v = acc[mi][ni][r];
          if (R) v += R[(long)row * OUT + col];
          Y[(long)row * OUT + col] = v;
        }
      }
    }
}

// t = delta * (out / ||out||_row) * lambda    (rows of length 64)
__global__ void norm_scale(const float* __restrict__ outc, const float* __restrict__ delta,
                           const int* __restrict__ lam, float* __restrict__ t) {
  __shared__ float red[64];
  int row = blockIdx.x, tid = threadIdx.x;
  float v = outc[(long)row * 64 + tid];
  red[tid] = v * v;
  __syncthreads();
  for (int s = 32; s > 0; s >>= 1) {
    if (tid < s) red[tid] += red[tid + s];
    __syncthreads();
  }
  float scale = (float)(*lam) * rsqrtf(red[0]);
  t[(long)row * 64 + tid] = delta[(long)row * 64 + tid] * v * scale;
}

extern "C" void kernel_launch(void* const* d_in, const int* in_sizes, int n_in,
                              void* d_out, int out_size, void* d_ws, size_t ws_size,
                              hipStream_t stream) {
  const float* latent = (const float*)d_in[0];   // (4096,1024)
  const float* delta  = (const float*)d_in[1];   // (4096,64)
  const float* cA = (const float*)d_in[2],  *sbA = (const float*)d_in[3],  *ssA = (const float*)d_in[4];
  const float* cB = (const float*)d_in[5],  *sbB = (const float*)d_in[6],  *ssB = (const float*)d_in[7];
  const float* cC = (const float*)d_in[8],  *sbC = (const float*)d_in[9],  *ssC = (const float*)d_in[10];
  const float* cE = (const float*)d_in[11], *sbE = (const float*)d_in[12], *ssE = (const float*)d_in[13];
  const int*   lam = (const int*)d_in[14];
  float* out = (float*)d_out;

  const int D = 1024, H = 2048, A = 64;
  size_t off = 0;
  auto alloc = [&](size_t bytes) -> void* {
    void* p = (char*)d_ws + off;
    off += (bytes + 255) & ~(size_t)255;
    return p;
  };
  unsigned short* Wa = (unsigned short*)alloc((size_t)D * FPAD * H * 2);
  unsigned short* Wb = (unsigned short*)alloc((size_t)H * FPAD * H * 2);
  unsigned short* Wc = (unsigned short*)alloc((size_t)H * FPAD * A * 2);
  unsigned short* We = (unsigned short*)alloc((size_t)A * FPAD * D * 2);
  unsigned short* Fb = (unsigned short*)alloc((size_t)BSZ * H * FPAD * 2); // reused
  float* h1   = (float*)alloc((size_t)BSZ * H * 4);
  float* h2   = (float*)alloc((size_t)BSZ * H * 4);
  float* out3 = (float*)alloc((size_t)BSZ * A * 4);
  float* tbuf = (float*)alloc((size_t)BSZ * A * 4);

  // 1) pack fused bf16 weights
  pack_w<<<((long)D * H + 255) / 256, 256, 0, stream>>>(cA, sbA, ssA, Wa, D, H);
  pack_w<<<((long)H * H + 255) / 256, 256, 0, stream>>>(cB, sbB, ssB, Wb, H, H);
  pack_w<<<((long)H * A + 255) / 256, 256, 0, stream>>>(cC, sbC, ssC, Wc, H, A);
  pack_w<<<((long)A * D + 255) / 256, 256, 0, stream>>>(cE, sbE, ssE, We, A, D);

  // 2) layer k1a: D -> H
  expand_feats<<<((long)BSZ * D + 255) / 256, 256, 0, stream>>>(latent, Fb, (long)BSZ * D);
  kan_gemm<<<dim3(BSZ / MT, (H + NT - 1) / NT), 256, 0, stream>>>(Fb, Wa, h1, nullptr, D * FPAD, H);

  // 3) layer k1b: H -> H
  expand_feats<<<((long)BSZ * H + 255) / 256, 256, 0, stream>>>(h1, Fb, (long)BSZ * H);
  kan_gemm<<<dim3(BSZ / MT, (H + NT - 1) / NT), 256, 0, stream>>>(Fb, Wb, h2, nullptr, H * FPAD, H);

  // 4) layer k1c: H -> A
  expand_feats<<<((long)BSZ * H + 255) / 256, 256, 0, stream>>>(h2, Fb, (long)BSZ * H);
  kan_gemm<<<dim3(BSZ / MT, (A + NT - 1) / NT), 256, 0, stream>>>(Fb, Wc, out3, nullptr, H * FPAD, A);

  // 5) row-normalize * lambda, multiply by delta
  norm_scale<<<BSZ, A, 0, stream>>>(out3, delta, lam, tbuf);

  // 6) embedding layer with residual add of latent_codes
  expand_feats<<<((long)BSZ * A + 255) / 256, 256, 0, stream>>>(tbuf, Fb, (long)BSZ * A);
  kan_gemm<<<dim3(BSZ / MT, (D + NT - 1) / NT), 256, 0, stream>>>(Fb, We, out, latent, A * FPAD, D);
}
